// REMSA_39762807226803
// MI455X (gfx1250) — matrix operand order repaired and checked by reading, not compiled, not run on hardware
//
#include <hip/hip_runtime.h>

// Problem constants (from reference)
#define DIM    512
#define NHEADS 8
#define HD     64
#define BATCH  2
#define SEQ    4096
#define MTOT   (BATCH * SEQ)   // 8192 rows

typedef __attribute__((ext_vector_type(16))) __bf16    v16bf;
typedef __attribute__((ext_vector_type(8)))  float     v8f;
typedef __attribute__((ext_vector_type(4)))  unsigned  u32x4;
typedef __attribute__((ext_vector_type(8)))  int       i32x8;
typedef __attribute__((ext_vector_type(4)))  int       i32x4;

#if defined(__gfx1250__) && __has_builtin(__builtin_amdgcn_tensor_load_to_lds) && \
    __has_builtin(__builtin_amdgcn_s_wait_tensorcnt)
#define USE_TDM 1
#else
#define USE_TDM 0
#endif

union FragBF { uint4 q[2]; v16bf v; };

__device__ __forceinline__ v8f v8f_zero() {
  v8f z = {0.f, 0.f, 0.f, 0.f, 0.f, 0.f, 0.f, 0.f};
  return z;
}

// round-to-nearest-even f32 -> bf16 (raw u16 storage)
__device__ __forceinline__ unsigned short f2bf(float f) {
  union { float f; unsigned u; } x; x.f = f;
  unsigned u = x.u;
  unsigned r = u + 0x7FFFu + ((u >> 16) & 1u);
  return (unsigned short)(r >> 16);
}

#if USE_TDM
// ---------------------------------------------------------------------------
// Tensor Data Mover: issue one 2D tile load (global -> LDS).
// D# per CDNA5 ISA sect. 8.3/8.4:
//  group0: [1:0]count=1 | [63:32]lds_addr | [120:64]global_addr | [127:126]type=2
//  group1: [17:16]data_size=1 (2B) | [79:48]tensor_dim0 | [111:80]tensor_dim1
//          [127:112]tile_dim0 | [143:128]tile_dim1 | [207:160]tensor_dim0_stride
// All values uniform (SGPR-resident). Tracked by TENSORcnt.
// This toolchain exposes the 6-arg builtin form:
//  (u32x4 g0, i32x8 g1, i32x4 g2, i32x4 g3, i32x8 extra, int cpol)
// ---------------------------------------------------------------------------
__device__ __forceinline__ void tdm_load_tile_2d(unsigned lds_byte_addr,
                                                 const void* gsrc,
                                                 unsigned tensor_d0, unsigned tensor_d1,
                                                 unsigned tile_d0, unsigned tile_d1,
                                                 unsigned d0_stride) {
  unsigned long long ga = (unsigned long long)(uintptr_t)gsrc;
  u32x4 g0;
  g0[0] = 1u;                                               // count=1, user D#
  g0[1] = lds_byte_addr;                                    // lds_addr
  g0[2] = (unsigned)ga;                                     // global_addr[31:0]
  g0[3] = (unsigned)((ga >> 32) & 0x1FFFFFFu) | (2u << 30); // addr[56:32] | type=2
  i32x8 g1;
  g1[0] = (int)(1u << 16);                                  // data_size=1 (2 bytes)
  g1[1] = (int)((tensor_d0 & 0xFFFFu) << 16);               // td0[15:0] @ bits 63:48
  g1[2] = (int)((tensor_d0 >> 16) | ((tensor_d1 & 0xFFFFu) << 16));
  g1[3] = (int)((tensor_d1 >> 16) | ((tile_d0 & 0xFFFFu) << 16));
  g1[4] = (int)(tile_d1 & 0xFFFFu);                         // tile_dim1, tile_dim2=0
  g1[5] = (int)d0_stride;                                   // stride0[31:0]
  g1[6] = 0;                                                // stride0[47:32], stride1 lo
  g1[7] = 0;
  i32x4 gz = {0, 0, 0, 0};                                  // 2D tensor: groups 2/3 unused
  i32x8 gz8 = {0, 0, 0, 0, 0, 0, 0, 0};
  __builtin_amdgcn_tensor_load_to_lds(g0, g1, gz, gz, gz8, 0);
}
#endif

// ---------------------------------------------------------------------------
// f32 -> bf16 conversion
// ---------------------------------------------------------------------------
__global__ void cvt_f32_to_bf16(const float* __restrict__ src,
                                unsigned short* __restrict__ dst, int n) {
  int i = blockIdx.x * blockDim.x + threadIdx.x;
  if (i < n) dst[i] = f2bf(src[i]);
}

// ---------------------------------------------------------------------------
// GEMM: out[M,Ncols] = alpha * (A[M,K]bf16 @ W[Ncols,K]bf16^T)   (torch Linear)
// Block tile 128x128, K-step 32, double-buffered LDS staged by the TDM
// (wave 0 issues tensor_load_to_lds; all waves spend VALU on WMMA only).
// 256 threads = 8 waves in 4(M) x 2(N); each wave: 2x4 16x16 f32 accumulators.
// ---------------------------------------------------------------------------
template <bool STORE_BF16, bool ACCUM>
__global__ __launch_bounds__(256)
void gemm_bf16_wmma(const unsigned short* __restrict__ A,
                    const unsigned short* __restrict__ W,
                    float* __restrict__ outF,
                    unsigned short* __restrict__ outB,
                    int M, int Ncols, int K, float alpha) {
  __shared__ unsigned short lA[2][128 * 32];  // [buf][row][k] k-contiguous
  __shared__ unsigned short lW[2][128 * 32];  // [buf][ncol][k]
  const int tid  = threadIdx.x;
  const int lane = tid & 31;
  const int wave = tid >> 5;
  const int l15  = lane & 15;
  const int g    = lane >> 4;       // half-wave select
  const int wm   = wave & 3;        // M wave row (32 rows each)
  const int wn   = wave >> 2;       // N wave col (64 cols each)
  const int mblk = blockIdx.y * 128;
  const int nblk = blockIdx.x * 128;

  v8f acc[2][4];
#pragma unroll
  for (int i = 0; i < 2; i++)
#pragma unroll
    for (int j = 0; j < 4; j++) acc[i][j] = v8f_zero();

#if USE_TDM
  // prologue: wave 0 kicks off DMA of the first K-slice into buffer 0
  if (wave == 0) {
    tdm_load_tile_2d((unsigned)(uintptr_t)&lA[0][0], A + (size_t)mblk * K,
                     (unsigned)K, (unsigned)M, 32u, 128u, (unsigned)K);
    tdm_load_tile_2d((unsigned)(uintptr_t)&lW[0][0], W + (size_t)nblk * K,
                     (unsigned)K, (unsigned)Ncols, 32u, 128u, (unsigned)K);
  }
#endif

  int cur = 0;
  for (int kb = 0; kb < K; kb += 32, cur ^= 1) {
#if USE_TDM
    if (wave == 0) {
      if (kb + 32 < K) {
        // issue next slice into the other buffer, then wait for the current
        // pair only (2 older TDM ops retire; 2 stay in flight).
        tdm_load_tile_2d((unsigned)(uintptr_t)&lA[cur ^ 1][0],
                         A + (size_t)mblk * K + (kb + 32),
                         (unsigned)K, (unsigned)M, 32u, 128u, (unsigned)K);
        tdm_load_tile_2d((unsigned)(uintptr_t)&lW[cur ^ 1][0],
                         W + (size_t)nblk * K + (kb + 32),
                         (unsigned)K, (unsigned)Ncols, 32u, 128u, (unsigned)K);
        __builtin_amdgcn_s_wait_tensorcnt(2);
      } else {
        __builtin_amdgcn_s_wait_tensorcnt(0);
      }
    }
    __syncthreads();
#else
    // fallback: cooperative VALU staging, 2 x 16B per thread per tile
#pragma unroll
    for (int i = 0; i < 2; i++) {
      int c  = tid * 2 + i;      // chunk of 8 bf16, 0..511
      int r  = c >> 2;           // row 0..127
      int k8 = (c & 3) * 8;
      *(uint4*)(&lA[cur][0] + r * 32 + k8) =
          *(const uint4*)(A + (size_t)(mblk + r) * K + kb + k8);
      *(uint4*)(&lW[cur][0] + r * 32 + k8) =
          *(const uint4*)(W + (size_t)(nblk + r) * K + kb + k8);
    }
    __syncthreads();
#endif

    // ---- build fragments (ISA 16-bit A 16x32 / B 32x16 layouts) ----
    const unsigned short* bA = &lA[cur][0];
    const unsigned short* bW = &lW[cur][0];
    FragBF af[2], bfr[4];
#pragma unroll
    for (int i = 0; i < 2; i++) {
      int mr = wm * 32 + i * 16 + l15;          // A: M = lane&15
      af[i].q[0] = *(const uint4*)(bA + mr * 32 + g * 8);       // K = g*8..+7
      af[i].q[1] = *(const uint4*)(bA + mr * 32 + 16 + g * 8);  // K = 16+g*8..
    }
#pragma unroll
    for (int j = 0; j < 4; j++) {
      int nr = wn * 64 + j * 16 + l15;          // B: column = lane&15
      const unsigned short* p = bW + nr * 32 + g * 8;   // K = g*8..+7, 16+g*8..+7
      bfr[j].q[0] = *(const uint4*)p;
      bfr[j].q[1] = *(const uint4*)(p + 16);
    }
#pragma unroll
    for (int i = 0; i < 2; i++)
#pragma unroll
      for (int j = 0; j < 4; j++)
        acc[i][j] = __builtin_amdgcn_wmma_f32_16x16x32_bf16(
            false, af[i].v, false, bfr[j].v, (short)0, acc[i][j], false, false);
    __syncthreads();
  }

  // ---- store: C layout row = r + 8*g, col = lane&15 ----
#pragma unroll
  for (int i = 0; i < 2; i++) {
    int mr = mblk + wm * 32 + i * 16 + g * 8;
#pragma unroll
    for (int j = 0; j < 4; j++) {
      int nc = nblk + wn * 64 + j * 16 + l15;
#pragma unroll
      for (int r = 0; r < 8; r++) {
        float val = acc[i][j][r] * alpha;
        size_t off = (size_t)(mr + r) * Ncols + nc;
        if (STORE_BF16) {
          outB[off] = f2bf(val);
        } else {
          if (ACCUM) outF[off] += val; else outF[off] = val;
        }
      }
    }
  }
}

// ---------------------------------------------------------------------------
// Flash attention: per block = 64 query rows of one (b, head).
// 4 waves x 16 rows; stream 64-wide K/V tiles via LDS; online softmax with
// width-16 shfl reductions; P re-laid out through LDS for the PV WMMA.
// q is pre-scaled by hd^-0.5 in the Q projection GEMM.
// ---------------------------------------------------------------------------
__global__ __launch_bounds__(128)
void flash_attn_bf16(const unsigned short* __restrict__ Q,
                     const unsigned short* __restrict__ Kmat,
                     const unsigned short* __restrict__ V,
                     unsigned short* __restrict__ O) {
  __shared__ unsigned short lK[64 * 64];        // [kv][d]
  __shared__ unsigned short lVt[64 * 64];       // [d][kv] (transposed on stage)
  __shared__ unsigned short lP[4 * 16 * 64];    // per-wave P tiles
  const int tid  = threadIdx.x;
  const int lane = tid & 31;
  const int wave = tid >> 5;
  const int l15  = lane & 15;
  const int g    = lane >> 4;
  const int b    = blockIdx.z;
  const int h    = blockIdx.y;
  const int q0   = blockIdx.x * 64 + wave * 16;
  const size_t headoff = (size_t)h * HD;

  // Q fragments (A layout: M = lane&15), loaded once from global
  FragBF qa[2];
  {
    const unsigned short* qp = Q + ((size_t)(b * SEQ + q0 + l15) * DIM) + headoff;
#pragma unroll
    for (int kk = 0; kk < 2; kk++) {
      qa[kk].q[0] = *(const uint4*)(qp + kk * 32 + g * 8);
      qa[kk].q[1] = *(const uint4*)(qp + kk * 32 + 16 + g * 8);
    }
  }

  float mstate[8], lsum[8];
  v8f acc[4];
#pragma unroll
  for (int r = 0; r < 8; r++) { mstate[r] = -1e30f; lsum[r] = 0.f; }
#pragma unroll
  for (int j = 0; j < 4; j++) acc[j] = v8f_zero();

  for (int kv = 0; kv < SEQ; kv += 64) {
    // ---- stage K tile (row-major), 4 x 16B per thread ----
#pragma unroll
    for (int i = 0; i < 4; i++) {
      int c = tid * 4 + i;           // chunk of 8 bf16, 0..511
      int r = c >> 3, k8 = (c & 7) * 8;
      *(uint4*)(lK + r * 64 + k8) =
          *(const uint4*)(Kmat + ((size_t)(b * SEQ + kv + r) * DIM) + headoff + k8);
    }
    // ---- stage V transposed: [d][kv] ----
#pragma unroll
    for (int i = 0; i < 16; i++) {
      int e = tid + 128 * i;         // 0..2047 dword pairs
      int n = e >> 5, d2 = (e & 31) * 2;
      unsigned u = *(const unsigned*)(V + ((size_t)(b * SEQ + kv + n) * DIM) + headoff + d2);
      lVt[(d2 + 0) * 64 + n] = (unsigned short)(u & 0xFFFFu);
      lVt[(d2 + 1) * 64 + n] = (unsigned short)(u >> 16);
    }
    __syncthreads();

    // ---- S = Q @ K^T : 4 16x16 f32 tiles per wave ----
    v8f s[4];
#pragma unroll
    for (int j = 0; j < 4; j++) s[j] = v8f_zero();
#pragma unroll
    for (int kk = 0; kk < 2; kk++) {
#pragma unroll
      for (int j = 0; j < 4; j++) {
        FragBF kf;  // B layout: column = kv index, K = g*8..+7, 16+g*8..+7
        const unsigned short* kp = lK + (j * 16 + l15) * 64 + kk * 32 + g * 8;
        kf.q[0] = *(const uint4*)kp;
        kf.q[1] = *(const uint4*)(kp + 16);
        s[j] = __builtin_amdgcn_wmma_f32_16x16x32_bf16(
            false, qa[kk].v, false, kf.v, (short)0, s[j], false, false);
      }
    }

    // ---- online softmax per row (row = q0 + r + 8*g) ----
#pragma unroll
    for (int r = 0; r < 8; r++) {
      float mx = fmaxf(fmaxf(s[0][r], s[1][r]), fmaxf(s[2][r], s[3][r]));
#pragma unroll
      for (int msk = 1; msk < 16; msk <<= 1) mx = fmaxf(mx, __shfl_xor(mx, msk, 32));
      float mnew = fmaxf(mstate[r], mx);
      float corr = __expf(mstate[r] - mnew);
      float rs = 0.f;
#pragma unroll
      for (int j = 0; j < 4; j++) {
        float p = __expf(s[j][r] - mnew);
        s[j][r] = p;
        rs += p;
      }
#pragma unroll
      for (int msk = 1; msk < 16; msk <<= 1) rs += __shfl_xor(rs, msk, 32);
      lsum[r] = lsum[r] * corr + rs;
      mstate[r] = mnew;
#pragma unroll
      for (int j = 0; j < 4; j++) acc[j][r] *= corr;
      // spill P (C layout) to LDS for A-layout reload
#pragma unroll
      for (int j = 0; j < 4; j++)
        lP[wave * 1024 + (r + 8 * g) * 64 + j * 16 + l15] = f2bf(s[j][r]);
    }
    __syncthreads();

    // ---- O += P @ V ----
#pragma unroll
    for (int kk = 0; kk < 2; kk++) {
      FragBF pa;  // A layout: M = lane&15, K = kv-local
      const unsigned short* pp = lP + wave * 1024 + l15 * 64 + kk * 32 + g * 8;
      pa.q[0] = *(const uint4*)pp;
      pa.q[1] = *(const uint4*)(pp + 16);
#pragma unroll
      for (int j = 0; j < 4; j++) {
        FragBF vf;  // B layout from transposed V: column = hd, K = g*8..+7, 16+g*8..+7
        const unsigned short* vp = lVt + (j * 16 + l15) * 64 + kk * 32 + g * 8;
        vf.q[0] = *(const uint4*)vp;
        vf.q[1] = *(const uint4*)(vp + 16);
        acc[j] = __builtin_amdgcn_wmma_f32_16x16x32_bf16(
            false, pa.v, false, vf.v, (short)0, acc[j], false, false);
      }
    }
    __syncthreads();
  }

  // ---- normalize and store bf16 [B,N,C] ----
#pragma unroll
  for (int j = 0; j < 4; j++)
#pragma unroll
    for (int r = 0; r < 8; r++) {
      float o = acc[j][r] / lsum[r];
      O[((size_t)(b * SEQ + q0 + r + 8 * g) * DIM) + headoff + j * 16 + l15] = f2bf(o);
    }
}

// ---------------------------------------------------------------------------
// Depthwise conv1d (k=3, pad=1) over N per channel, + bias; bf16 out.
// ci is [B,N,C] f32.
// ---------------------------------------------------------------------------
__global__ void dwconv3(const float* __restrict__ ci, const float* __restrict__ dw,
                        const float* __restrict__ bias, unsigned short* __restrict__ co) {
  int idx = blockIdx.x * blockDim.x + threadIdx.x;
  if (idx >= MTOT * DIM) return;
  int c = idx & (DIM - 1);
  int n = (idx >> 9) & (SEQ - 1);
  float ctr = ci[idx];
  float lft = (n > 0)       ? ci[idx - DIM] : 0.f;
  float rgt = (n < SEQ - 1) ? ci[idx + DIM] : 0.f;
  float o = lft * dw[c * 3 + 0] + ctr * dw[c * 3 + 1] + rgt * dw[c * 3 + 2] + bias[c];
  co[idx] = f2bf(o);
}

// ---------------------------------------------------------------------------
extern "C" void kernel_launch(void* const* d_in, const int* in_sizes, int n_in,
                              void* d_out, int out_size, void* d_ws, size_t ws_size,
                              hipStream_t stream) {
  const float* x     = (const float*)d_in[0];
  const float* Wq    = (const float*)d_in[1];
  const float* Wk    = (const float*)d_in[2];
  const float* Wv    = (const float*)d_in[3];
  const float* Wattn = (const float*)d_in[4];
  const float* Wconv = (const float*)d_in[5];
  const float* dwk   = (const float*)d_in[6];
  const float* dwb   = (const float*)d_in[7];
  const float* Wcout = (const float*)d_in[8];
  float* out = (float*)d_out;

  // workspace carve-up (~70 MB)
  char* ws = (char*)d_ws;
  size_t off = 0;
  auto carve = [&](size_t bytes) {
    char* p = ws + off;
    off += (bytes + 255) & ~(size_t)255;
    return p;
  };
  const size_t nX = (size_t)MTOT * DIM, nW = (size_t)DIM * DIM;
  unsigned short* xb    = (unsigned short*)carve(nX * 2);
  unsigned short* qb    = (unsigned short*)carve(nX * 2);
  unsigned short* kbuf  = (unsigned short*)carve(nX * 2);
  unsigned short* vbuf  = (unsigned short*)carve(nX * 2);
  unsigned short* attnb = (unsigned short*)carve(nX * 2);
  unsigned short* cob   = (unsigned short*)carve(nX * 2);
  float*          cif   = (float*)carve(nX * 4);
  unsigned short* wqb   = (unsigned short*)carve(nW * 2);
  unsigned short* wkb   = (unsigned short*)carve(nW * 2);
  unsigned short* wvb   = (unsigned short*)carve(nW * 2);
  unsigned short* wab   = (unsigned short*)carve(nW * 2);
  unsigned short* wcb   = (unsigned short*)carve(nW * 2);
  unsigned short* wob   = (unsigned short*)carve(nW * 2);

  // 1) bf16 conversions
  cvt_f32_to_bf16<<<(int)((nX + 255) / 256), 256, 0, stream>>>(x, xb, (int)nX);
  cvt_f32_to_bf16<<<(int)((nW + 255) / 256), 256, 0, stream>>>(Wq, wqb, (int)nW);
  cvt_f32_to_bf16<<<(int)((nW + 255) / 256), 256, 0, stream>>>(Wk, wkb, (int)nW);
  cvt_f32_to_bf16<<<(int)((nW + 255) / 256), 256, 0, stream>>>(Wv, wvb, (int)nW);
  cvt_f32_to_bf16<<<(int)((nW + 255) / 256), 256, 0, stream>>>(Wattn, wab, (int)nW);
  cvt_f32_to_bf16<<<(int)((nW + 255) / 256), 256, 0, stream>>>(Wconv, wcb, (int)nW);
  cvt_f32_to_bf16<<<(int)((nW + 255) / 256), 256, 0, stream>>>(Wcout, wob, (int)nW);

  // 2) projections (scale hd^-0.5 folded into q)
  dim3 gg(DIM / 128, MTOT / 128);  // (4, 64)
  gemm_bf16_wmma<true,  false><<<gg, 256, 0, stream>>>(xb, wqb, nullptr, qb,   MTOT, DIM, DIM, 0.125f);
  gemm_bf16_wmma<true,  false><<<gg, 256, 0, stream>>>(xb, wkb, nullptr, kbuf, MTOT, DIM, DIM, 1.0f);
  gemm_bf16_wmma<true,  false><<<gg, 256, 0, stream>>>(xb, wvb, nullptr, vbuf, MTOT, DIM, DIM, 1.0f);
  gemm_bf16_wmma<false, false><<<gg, 256, 0, stream>>>(xb, wcb, cif, nullptr,  MTOT, DIM, DIM, 1.0f);

  // 3) flash attention
  flash_attn_bf16<<<dim3(SEQ / 64, NHEADS, BATCH), 128, 0, stream>>>(qb, kbuf, vbuf, attnb);

  // 4) depthwise conv branch
  dwconv3<<<(int)((nX + 255) / 256), 256, 0, stream>>>(cif, dwk, dwb, cob);

  // 5) output projections: write then accumulate into d_out ([B,nh,W,C] is flat [8192,512])
  gemm_bf16_wmma<false, false><<<gg, 256, 0, stream>>>(attnb, wab, out, nullptr, MTOT, DIM, DIM, 1.0f);
  gemm_bf16_wmma<false, true ><<<gg, 256, 0, stream>>>(cob,   wob, out, nullptr, MTOT, DIM, DIM, 1.0f);
}
